// RelationalGraphletConvolution_4372276708025
// MI455X (gfx1250) — matrix operand order repaired and checked by reading, not compiled, not run on hardware
//
#include <hip/hip_runtime.h>

// Problem constants (match reference)
#define NOBJ 32
#define RELD 16
#define NFLT 16
#define NGRP 4960          // C(32,3)
#define NQ   512
#define NB   32
#define KDIM 144           // 3*3*16, padded to 160 = 5 chunks of 32
#define NC   (NGRP / 32)   // 155 K-chunks in GEMM2

typedef __attribute__((ext_vector_type(16))) _Float16 v16h;
typedef __attribute__((ext_vector_type(8)))  _Float16 v8h;
typedef __attribute__((ext_vector_type(8)))  float    v8f;

static __device__ __forceinline__ v8h ld_v8h(const _Float16* p) {
    return *reinterpret_cast<const v8h*>(p);
}
static __device__ __forceinline__ v8f ld_v8f(const float* p) {
    return *reinterpret_cast<const v8f*>(p);
}
// Load 16 B from LDS at a raw byte offset through an addrspace(3) pointer:
// guarantees ds_load_b128 and matches the raw offsets used by the async-copy asm.
static __device__ __forceinline__ v8h ld_lds16(uint32_t byte_off) {
    return *(const __attribute__((address_space(3))) v8h*)(uintptr_t)byte_off;
}

// ---------------------------------------------------------------------------
// Kernel 0: regenerate GROUP_IDX (lexicographic combinations of 32 choose 3)
// ---------------------------------------------------------------------------
__global__ void build_triples(int4* __restrict__ trips) {
    int g = blockIdx.x * blockDim.x + threadIdx.x;
    if (g >= NGRP) return;
    int r = g, i = 0;
    for (;;) { int c = (31 - i) * (30 - i) / 2; if (r < c) break; r -= c; ++i; }
    int j = i + 1;
    for (;;) { int c = 31 - j; if (r < c) break; r -= c; ++j; }
    int k = j + 1 + r;
    trips[g] = make_int4(i, j, k, 0);
}

// ---------------------------------------------------------------------------
// Kernel 1: sp = softplus(logits); w[q,g] = prod sp[q, trip]; softmax over g
// One block per query q. Stores norm_w as f16 [NQ][NGRP] (A-matrix of GEMM2).
// ---------------------------------------------------------------------------
__global__ __launch_bounds__(128)
void weights_softmax(const float* __restrict__ logits,
                     const int4*  __restrict__ trips,
                     _Float16*    __restrict__ nw) {
    __shared__ float sp[NOBJ];
    __shared__ float buf[NGRP];
    __shared__ float red[128];
    const int q = blockIdx.x;
    const int tid = threadIdx.x;

    if (tid < NOBJ) {
        float x = logits[q * NOBJ + tid];
        sp[tid] = (x > 15.f) ? x : log1pf(expf(x));   // stable softplus
    }
    __syncthreads();

    float lmax = -3.4e38f;
    for (int g = tid; g < NGRP; g += 128) {
        int4 t = trips[g];
        float w = sp[t.x] * sp[t.y] * sp[t.z];
        buf[g] = w;
        lmax = fmaxf(lmax, w);
    }
    red[tid] = lmax; __syncthreads();
    for (int s = 64; s > 0; s >>= 1) {
        if (tid < s) red[tid] = fmaxf(red[tid], red[tid + s]);
        __syncthreads();
    }
    const float wmax = red[0];
    __syncthreads();

    float lsum = 0.f;
    for (int g = tid; g < NGRP; g += 128) {
        float e = expf(buf[g] - wmax);
        buf[g] = e;
        lsum += e;
    }
    red[tid] = lsum; __syncthreads();
    for (int s = 64; s > 0; s >>= 1) {
        if (tid < s) red[tid] += red[tid + s];
        __syncthreads();
    }
    const float inv = 1.f / red[0];
    for (int g = tid; g < NGRP; g += 128)
        nw[(size_t)q * NGRP + g] = (_Float16)(buf[g] * inv);
}

// ---------------------------------------------------------------------------
// Kernel 2 (GEMM1, WMMA): rel_conv[b,g,f] = sum_{i,j,d} in[b,ti,tj,d]*flt[f,i,j,d]
// One wave per 16-row tile of the flattened (b,g) dimension. K=144 -> 5 chunks.
// Output stored TRANSPOSED f16: rc_t[(b*16+f)][g]  -> packed v8h stores, and
// contiguous-K B-fragments for GEMM2.
// ---------------------------------------------------------------------------
__global__ __launch_bounds__(256)
void gemm1_relconv(const float* __restrict__ inputs,
                   const float* __restrict__ filters,
                   const int4*  __restrict__ trips,
                   _Float16*    __restrict__ rc_t) {
    const int lane  = threadIdx.x & 31;
    const int wave  = threadIdx.x >> 5;
    const int tile  = blockIdx.x * 8 + wave;       // 9920 tiles total
    const int row16 = tile * 16;                   // NGRP % 16 == 0 -> one b per tile
    const int b     = row16 / NGRP;
    const int gbase = row16 % NGRP;
    const bool hi   = lane >= 16;
    const int m     = lane & 15;                   // A-matrix row (this lane)
    const int f     = lane & 15;                   // B-matrix column (this lane)
    const int g     = gbase + m;

    const int4 t = trips[g];
    const float* inb  = inputs + (size_t)b * NOBJ * NOBJ * RELD;
    const float* fltf = filters + f * KDIM;
    const int kbA = hi ? 8 : 0;                    // A lane K-base (16-bit A 16x32 layout)
    const int kbB = hi ? 8 : 0;                    // B lane K-base (16-bit B 32x16 layout)

    v8f acc = {};
#pragma unroll
    for (int c = 0; c < 5; ++c) {
        // ---- A fragment: two contiguous 8-float runs (d-dimension) ----
        v16h a;
#pragma unroll
        for (int half = 0; half < 2; ++half) {
            const int kk0 = 32 * c + kbA + 16 * half;   // run start (multiple of 8)
            v8f av = {};
            if (kk0 < KDIM) {
                const int i  = kk0 / 48;
                const int rm = kk0 % 48;
                const int j  = rm / 16;
                const int d0 = rm % 16;
                const int ti = (i == 0) ? t.x : (i == 1) ? t.y : t.z;
                const int tj = (j == 0) ? t.x : (j == 1) ? t.y : t.z;
                av = ld_v8f(&inb[(ti * NOBJ + tj) * RELD + d0]);
            }
#pragma unroll
            for (int e = 0; e < 8; ++e) a[half * 8 + e] = (_Float16)av[e];
        }
        // ---- B fragment: contiguous runs of the filter row ----
        v16h bm;
#pragma unroll
        for (int half = 0; half < 2; ++half) {
            const int kk0 = 32 * c + kbB + 16 * half;
            v8f bv = {};
            if (kk0 < KDIM) bv = ld_v8f(&fltf[kk0]);
#pragma unroll
            for (int e = 0; e < 8; ++e) bm[half * 8 + e] = (_Float16)bv[e];
        }
        acc = __builtin_amdgcn_wmma_f32_16x16x32_f16(false, a, false, bm,
                                                     (short)0, acc, false, false);
    }

    // D layout: VGPR r -> M = r + 8*hi, N = lane&15. M maps to g (contiguous!)
    v8h st;
#pragma unroll
    for (int r = 0; r < 8; ++r) st[r] = (_Float16)acc[r];
    _Float16* dst = rc_t + (size_t)(b * NFLT + f) * NGRP + gbase + (hi ? 8 : 0);
    *reinterpret_cast<v8h*>(dst) = st;
}

// ---------------------------------------------------------------------------
// Kernel 3 (GEMM2, WMMA + async-to-LDS): out[b,q,f] = sum_g nw[q,g]*rc_t[b*16+f][g]
// Block = 4 waves sharing one batch-group (4 batches). Each wave async-stages
// its batch's 32x16 f16 B-tile into LDS (GLOBAL_LOAD_ASYNC_TO_LDS_B128,
// ASYNCcnt, double-buffered); all waves consume all 4 tiles via ds_load_b128
// through explicit addrspace(3) pointers. Wave w computes query-tile
// qt_base+w across all 4 batches (4 accumulators).
// ---------------------------------------------------------------------------
__global__ __launch_bounds__(128)
void gemm2_out(const _Float16* __restrict__ nw,
               const _Float16* __restrict__ rc_t,
               float*          __restrict__ out) {
    // smem[buf][bb][f*32 + e] halfs; single shared object => LDS byte offset 0
    __shared__ _Float16 smem[2][4][512];           // 8 KB

    const int lane  = threadIdx.x & 31;
    const int wave  = threadIdx.x >> 5;            // 0..3
    const int bg    = blockIdx.x >> 3;             // 0..7 (batch group of 4)
    const int qtile = (blockIdx.x & 7) * 4 + wave; // 0..31
    const bool hi   = lane >= 16;
    const int m     = lane & 15;
    const int q     = qtile * 16 + m;
    const int kbA   = hi ? 16 : 0;
    const int kbB   = hi ? 16 : 0;
    const _Float16* arow = nw + (size_t)q * NGRP;

    // Keep the LDS allocation alive with a real store, and order it before the
    // first async write (syncthreads waits DScnt before the barrier).
    smem[0][0][threadIdx.x] = (_Float16)0;
    __syncthreads();

    // Async stage: this wave copies batch (bg*4+wave)'s 32g x 16f tile (1 KB).
    // Lane covers 32 B: f = lane>>1, ep = (lane&1)*16 halfs; two b128 ops,
    // INST_OFFSET is added to BOTH the LDS and global addresses.
    const int sf = lane >> 1;
    const int sep = (lane & 1) * 16;
    const _Float16* gbase_ptr =
        rc_t + (size_t)((bg * 4 + wave) * NFLT + sf) * NGRP + sep;
    const uint32_t lbase = (uint32_t)(((wave * 512) + sf * 32 + sep) * sizeof(_Float16));

#define STAGE(cc, bufsel)                                                      \
    do {                                                                       \
        const _Float16* gs = gbase_ptr + 32 * (cc);                            \
        const uint32_t la = lbase + (uint32_t)((bufsel) * 4 * 512 * 2);        \
        asm volatile("global_load_async_to_lds_b128 %0, %1, off"               \
                     :: "v"(la), "v"(gs) : "memory");                          \
        asm volatile("global_load_async_to_lds_b128 %0, %1, off offset:16"     \
                     :: "v"(la), "v"(gs) : "memory");                          \
    } while (0)

    v8f acc[4] = {};
    STAGE(0, 0);
    for (int c = 0; c < NC; ++c) {
        // wait own async stage of chunk c, then block-wide handshake
        asm volatile("s_wait_asynccnt 0x0" ::: "memory");
        __syncthreads();
        if (c + 1 < NC) STAGE(c + 1, (c + 1) & 1);

        // A fragment (norm_w): two 16B global loads + prefetch-ahead
        __builtin_prefetch(arow + 32 * (c + 4), 0, 1);
        v16h a;
        {
            v8h lo  = ld_v8h(arow + 32 * c + kbA);
            v8h hi8 = ld_v8h(arow + 32 * c + kbA + 8);
#pragma unroll
            for (int e = 0; e < 8; ++e) { a[e] = lo[e]; a[8 + e] = hi8[e]; }
        }
        // B fragments from LDS (two ds_load_b128 each), one WMMA per batch
        const int buf = c & 1;
#pragma unroll
        for (int bb = 0; bb < 4; ++bb) {
            const uint32_t soff =
                (uint32_t)(((((buf * 4 + bb) * 512) + (lane & 15) * 32 + kbB)) *
                           sizeof(_Float16));
            v8h lo  = ld_lds16(soff);
            v8h hi8 = ld_lds16(soff + 16);
            v16h bm;
#pragma unroll
            for (int e = 0; e < 8; ++e) { bm[e] = lo[e]; bm[8 + e] = hi8[e]; }
            acc[bb] = __builtin_amdgcn_wmma_f32_16x16x32_f16(false, a, false, bm,
                                                             (short)0, acc[bb],
                                                             false, false);
        }
    }
#undef STAGE

#pragma unroll
    for (int bb = 0; bb < 4; ++bb) {
        const int b = bg * 4 + bb;
#pragma unroll
        for (int r = 0; r < 8; ++r) {
            const int qq = qtile * 16 + r + (hi ? 8 : 0);
            out[((size_t)b * NQ + qq) * NFLT + (lane & 15)] = acc[bb][r];
        }
    }
}

// ---------------------------------------------------------------------------
extern "C" void kernel_launch(void* const* d_in, const int* in_sizes, int n_in,
                              void* d_out, int out_size, void* d_ws, size_t ws_size,
                              hipStream_t stream) {
    const float* inputs  = (const float*)d_in[0];   // (32,32,32,16) f32
    const float* logits  = (const float*)d_in[1];   // (512,32)      f32
    const float* filters = (const float*)d_in[2];   // (16,3,3,16)   f32
    float* out = (float*)d_out;                     // (32,512,16)   f32

    char* ws = (char*)d_ws;
    int4*     trips = (int4*)ws;                                   //  79,360 B
    _Float16* nw    = (_Float16*)(ws + (80u << 10));               // 5,079,040 B
    _Float16* rc_t  = (_Float16*)(ws + (80u << 10) + (5u << 20));  // 5,079,040 B

    build_triples<<<(NGRP + 255) / 256, 256, 0, stream>>>(trips);
    weights_softmax<<<NQ, 128, 0, stream>>>(logits, trips, nw);
    gemm1_relconv<<<(NB * NGRP / 16) / 8, 256, 0, stream>>>(inputs, filters, trips, rc_t);
    gemm2_out<<<64, 128, 0, stream>>>(nw, rc_t, out);
}
